// Three_Dim_LSTM_22935125360795
// MI455X (gfx1250) — matrix operand order repaired and checked by reading, not compiled, not run on hardware
//
#include <hip/hip_runtime.h>
#include <hip/hip_bf16.h>

// ---------------------------------------------------------------------------
// Types for CDNA5 WMMA
// ---------------------------------------------------------------------------
typedef __attribute__((ext_vector_type(16))) __bf16 v16bf;
typedef __attribute__((ext_vector_type(8)))  __bf16 v8bf;
typedef __attribute__((ext_vector_type(8)))  float  v8f;

// Problem constants (from reference)
constexpr int BATCH = 4096;
constexpr int NH    = 128;
constexpr int KX    = 1024;
constexpr int NL    = NH * 64;   // 8192
constexpr int VOX   = 64;        // 4*4*4
constexpr int GK    = 128;       // K elements staged in LDS per stage

__device__ __forceinline__ v16bf cat8(v8bf lo, v8bf hi) {
  return __builtin_shufflevector(lo, hi, 0,1,2,3,4,5,6,7,
                                          8,9,10,11,12,13,14,15);
}

__device__ __forceinline__ float sigmoidf_(float x) {
  return 1.0f / (1.0f + __expf(-x));
}

__device__ __forceinline__ v8f wmma_bf16(v16bf a, v16bf b, v8f c) {
  return __builtin_amdgcn_wmma_f32_16x16x32_bf16(
      false, a, false, b, (short)0, c, false, false);
}

// Async 16B global -> LDS copy (gfx1250, tracked by ASYNCcnt).
__device__ __forceinline__ void async_copy16(const __bf16* gsrc, __bf16* ldst) {
  unsigned int loff = (unsigned int)(unsigned long long)ldst;
  asm volatile("global_load_async_to_lds_b128 %0, %1, off"
               :: "v"(loff), "v"(gsrc) : "memory");
}

__device__ __forceinline__ void wait_async_and_sync() {
  asm volatile("s_wait_asynccnt 0" ::: "memory");
  __syncthreads();
}

// LDS fragment loads: lane half h takes k = ks + 16h + s (contiguous order, same for A and B).
__device__ __forceinline__ v16bf ldsA(const __bf16* s, int row, int ks, int h, int ld) {
  return *(const v16bf*)(s + row * ld + ks + h * 16);
}
__device__ __forceinline__ v16bf ldsB(const __bf16* s, int row, int ks, int h, int ld) {
  return *(const v16bf*)(s + row * ld + ks + h * 16);
}

// ---------------------------------------------------------------------------
// Packing kernels
// ---------------------------------------------------------------------------
__global__ void pack_bf16(const float* __restrict__ in,
                          __bf16* __restrict__ out, long n) {
  long i = (long)blockIdx.x * blockDim.x + threadIdx.x;
  if (i < n) out[i] = (__bf16)in[i];
}

// W [O=128][I=128][27] f32  ->  Wp [27][128(o)][128(i)] bf16
__global__ void pack_w(const float* __restrict__ W, __bf16* __restrict__ Wp) {
  int idx = blockIdx.x * blockDim.x + threadIdx.x;   // n = 27*128*128
  if (idx >= 27 * NH * NH) return;
  int i = idx & 127;
  int o = (idx >> 7) & 127;
  int t = idx >> 14;
  Wp[idx] = (__bf16)W[((size_t)o * NH + i) * 27 + t];
}

// hidden [b][c][p] f32 -> hp [p][b][c] bf16
__global__ void pack_h(const float* __restrict__ H, __bf16* __restrict__ hp) {
  long idx = (long)blockIdx.x * blockDim.x + threadIdx.x;  // 64*4096*128
  if (idx >= (long)VOX * BATCH * NH) return;
  int c = idx & 127;
  int b = (idx >> 7) & 4095;
  int p = (int)(idx >> 19);
  hp[idx] = (__bf16)H[((size_t)b * NH + c) * VOX + p];
}

// ---------------------------------------------------------------------------
// Double-buffered LDS-staged bf16 GEMM: C[M][N] = A[M][K] @ Bc[N][K]^T.
// Block = 256 threads (8 waves), block tile 64M x 128N, K staged 128/iter via
// async-to-LDS into buf[s&1] while buf[(s+1)&1] is being filled.  One barrier
// per stage; s_wait_asynccnt 0 covers copies issued a full stage earlier.
// Wave (wr, wc) register-blocks 2x2 16x16 tiles.
// ---------------------------------------------------------------------------
__global__ __launch_bounds__(256)
void gemm_bf16_lds(const __bf16* __restrict__ A,   // [M][K]
                   const __bf16* __restrict__ Bc,  // [N][K]
                   float* __restrict__ C,          // [M][N]
                   int M, int N, int K) {
  __shared__ __align__(128) __bf16 As[2][64 * GK];    // 2 x 16 KB
  __shared__ __align__(128) __bf16 Bs[2][128 * GK];   // 2 x 32 KB

  const int tid  = threadIdx.x;
  const int lane = tid & 31;
  const int wave = tid >> 5;
  const int h    = lane >> 4;
  const int mn   = lane & 15;
  const int wr   = wave >> 2;           // 0..1 : 32-row strip
  const int wc   = wave & 3;            // 0..3 : 32-col strip
  const int row0 = blockIdx.y * 64;
  const int col0 = blockIdx.x * 128;

  auto stage = [&](int buf, int k0) {
    for (int i = tid; i < (64 * GK) / 8; i += 256) {      // 4 chunks/thread
      int r = i >> 4, c = i & 15;
      async_copy16(A + (size_t)(row0 + r) * K + k0 + c * 8,
                   &As[buf][r * GK + c * 8]);
    }
    for (int i = tid; i < (128 * GK) / 8; i += 256) {     // 8 chunks/thread
      int r = i >> 4, c = i & 15;
      async_copy16(Bc + (size_t)(col0 + r) * K + k0 + c * 8,
                   &Bs[buf][r * GK + c * 8]);
    }
  };

  v8f acc[2][2] = {};
  const int nstage = K / GK;
  stage(0, 0);

  for (int s = 0; s < nstage; ++s) {
    const int cur = s & 1;
    wait_async_and_sync();                 // buf[cur] ready; prior reads done
    if (s + 1 < nstage) stage(cur ^ 1, (s + 1) * GK);

#pragma unroll
    for (int ks = 0; ks < GK; ks += 32) {
      v16bf a0 = ldsA(As[cur], wr * 32 + mn,      ks, h, GK);
      v16bf a1 = ldsA(As[cur], wr * 32 + 16 + mn, ks, h, GK);
      v16bf b0 = ldsB(Bs[cur], wc * 32 + mn,      ks, h, GK);
      v16bf b1 = ldsB(Bs[cur], wc * 32 + 16 + mn, ks, h, GK);
      acc[0][0] = wmma_bf16(a0, b0, acc[0][0]);
      acc[0][1] = wmma_bf16(a0, b1, acc[0][1]);
      acc[1][0] = wmma_bf16(a1, b0, acc[1][0]);
      acc[1][1] = wmma_bf16(a1, b1, acc[1][1]);
    }
  }

#pragma unroll
  for (int mi = 0; mi < 2; ++mi)
#pragma unroll
    for (int ni = 0; ni < 2; ++ni) {
      float* crow = C + (size_t)(row0 + wr * 32 + mi * 16) * N
                      + col0 + wc * 32 + ni * 16;
#pragma unroll
      for (int v = 0; v < 8; ++v)
        crow[(size_t)(v + 8 * h) * N + mn] = acc[mi][ni][v];
    }
}

// ---------------------------------------------------------------------------
// 3x3x3 conv on a 4x4x4 grid as a tap-accumulated GEMM:
//   out[p][b][o] = sum_t sum_i W[t][o][i] * hp[q(p,t)][b][i]
// Valid taps are a uniform-per-block bitmask (whole block shares p), so the
// double-buffered pipeline is a uniform popcount/ffs loop.  One LDS stage per
// valid tap (K=128); accumulators stay in registers across all taps.
// ---------------------------------------------------------------------------
__device__ __forceinline__ int tap_q(int p, int t) {
  int pz = p >> 4, py = (p >> 2) & 3, px = p & 3;
  int qz = pz + t / 9 - 1;
  int qy = py + (t / 3) % 3 - 1;
  int qx = px + t % 3 - 1;
  return (qz * 4 + qy) * 4 + qx;
}

__global__ __launch_bounds__(256)
void conv_gemm_lds(const __bf16* __restrict__ hp,  // [64(q)][4096(b)][128(i)]
                   const __bf16* __restrict__ Wp,  // [27(t)][128(o)][128(i)]
                   float* __restrict__ out) {      // [64(p)][4096(b)][128(o)]
  __shared__ __align__(128) __bf16 As[2][64 * NH];    // 2 x 16 KB
  __shared__ __align__(128) __bf16 Bs[2][NH * NH];    // 2 x 32 KB

  const int tid  = threadIdx.x;
  const int lane = tid & 31;
  const int wave = tid >> 5;
  const int h    = lane >> 4;
  const int mn   = lane & 15;
  const int wr   = wave >> 2;
  const int wc   = wave & 3;
  const int row0 = blockIdx.x * 64;   // batch tile
  const int p    = blockIdx.y;        // output voxel
  const int pz = p >> 4, py = (p >> 2) & 3, px = p & 3;

  // Uniform valid-tap bitmask
  unsigned mask = 0;
  for (int t = 0; t < 27; ++t) {
    int qz = pz + t / 9 - 1;
    int qy = py + (t / 3) % 3 - 1;
    int qx = px + t % 3 - 1;
    if ((unsigned)qz < 4u && (unsigned)qy < 4u && (unsigned)qx < 4u)
      mask |= 1u << t;
  }

  auto stage = [&](int buf, int t) {
    int q = tap_q(p, t);
    for (int i = tid; i < (64 * NH) / 8; i += 256) {
      int r = i >> 4, c = i & 15;
      async_copy16(hp + ((size_t)q * BATCH + row0 + r) * NH + c * 8,
                   &As[buf][r * NH + c * 8]);
    }
    for (int i = tid; i < (NH * NH) / 8; i += 256) {
      int r = i >> 4, c = i & 15;
      async_copy16(Wp + ((size_t)t * NH + r) * NH + c * 8,
                   &Bs[buf][r * NH + c * 8]);
    }
  };

  v8f acc[2][2] = {};
  const int ntap = __builtin_popcount(mask);
  int tnext = __builtin_ffs(mask) - 1;
  mask &= mask - 1;
  stage(0, tnext);

  for (int s = 0; s < ntap; ++s) {
    const int cur = s & 1;
    wait_async_and_sync();                 // buf[cur] ready; prior reads done
    if (mask) {
      tnext = __builtin_ffs(mask) - 1;
      mask &= mask - 1;
      stage(cur ^ 1, tnext);
    }

#pragma unroll
    for (int ks = 0; ks < NH; ks += 32) {
      v16bf a0 = ldsA(As[cur], wr * 32 + mn,      ks, h, NH);
      v16bf a1 = ldsA(As[cur], wr * 32 + 16 + mn, ks, h, NH);
      v16bf b0 = ldsB(Bs[cur], wc * 32 + mn,      ks, h, NH);
      v16bf b1 = ldsB(Bs[cur], wc * 32 + 16 + mn, ks, h, NH);
      acc[0][0] = wmma_bf16(a0, b0, acc[0][0]);
      acc[0][1] = wmma_bf16(a0, b1, acc[0][1]);
      acc[1][0] = wmma_bf16(a1, b0, acc[1][0]);
      acc[1][1] = wmma_bf16(a1, b1, acc[1][1]);
    }
  }

#pragma unroll
  for (int mi = 0; mi < 2; ++mi)
#pragma unroll
    for (int ni = 0; ni < 2; ++ni) {
      float* crow = out + ((size_t)p * BATCH + row0 + wr * 32 + mi * 16) * NH
                        + wc * 32 + ni * 16;
#pragma unroll
      for (int v = 0; v < 8; ++v)
        crow[(size_t)(v + 8 * h) * NH + mn] = acc[mi][ni][v];
    }
}

// ---------------------------------------------------------------------------
// Gate kernel: z = sigmoid(convZ + linZ + bz);  r = sigmoid(convR + linR + br)
// Emits z (f32, conv layout) and r*h already bf16-packed in conv-input layout.
// ---------------------------------------------------------------------------
__global__ void gate_zr(const float* __restrict__ convZ,  // [p][b][c]
                        const float* __restrict__ convR,
                        const float* __restrict__ linZ,   // [b][c*64+p]
                        const float* __restrict__ linR,
                        const float* __restrict__ bz,     // [c*64+p]
                        const float* __restrict__ br,
                        const float* __restrict__ hidden, // [b][c][p]
                        float* __restrict__ zbuf,         // [p][b][c]
                        __bf16* __restrict__ rhp) {       // [p][b][c] bf16
  long idx = (long)blockIdx.x * blockDim.x + threadIdx.x;
  if (idx >= (long)VOX * BATCH * NH) return;
  int c = idx & 127;
  int b = (idx >> 7) & 4095;
  int p = (int)(idx >> 19);
  size_t lc = (size_t)b * NL + c * VOX + p;
  int    bi = c * VOX + p;
  float zv = sigmoidf_(convZ[idx] + linZ[lc] + bz[bi]);
  float rv = sigmoidf_(convR[idx] + linR[lc] + br[bi]);
  float hv = hidden[((size_t)b * NH + c) * VOX + p];
  zbuf[idx] = zv;
  rhp[idx]  = (__bf16)(rv * hv);
}

// out = (1-z)*h + z*tanh(convM + linM + bm)
__global__ void final_gate(const float* __restrict__ convM,  // [p][b][c]
                           const float* __restrict__ linM,
                           const float* __restrict__ bm,
                           const float* __restrict__ hidden,
                           const float* __restrict__ zbuf,
                           float* __restrict__ out) {        // [b][c][p]
  long idx = (long)blockIdx.x * blockDim.x + threadIdx.x;
  if (idx >= (long)VOX * BATCH * NH) return;
  int c = idx & 127;
  int b = (idx >> 7) & 4095;
  int p = (int)(idx >> 19);
  size_t lc = (size_t)b * NL + c * VOX + p;
  int    bi = c * VOX + p;
  float mv = tanhf(convM[idx] + linM[lc] + bm[bi]);
  float zv = zbuf[idx];
  size_t ho = ((size_t)b * NH + c) * VOX + p;
  float hv = hidden[ho];
  out[ho] = (1.0f - zv) * hv + zv * mv;
}

// ---------------------------------------------------------------------------
// Launcher
// ---------------------------------------------------------------------------
extern "C" void kernel_launch(void* const* d_in, const int* in_sizes, int n_in,
                              void* d_out, int out_size, void* d_ws,
                              size_t ws_size, hipStream_t stream) {
  (void)in_sizes; (void)n_in; (void)out_size; (void)ws_size;

  const float* x      = (const float*)d_in[0];
  const float* hidden = (const float*)d_in[1];
  const float* Wz = (const float*)d_in[2];
  const float* Wr = (const float*)d_in[3];
  const float* Wm = (const float*)d_in[4];
  const float* bz = (const float*)d_in[5];
  const float* br = (const float*)d_in[6];
  const float* bm = (const float*)d_in[7];
  const float* Lz = (const float*)d_in[8];
  const float* Lr = (const float*)d_in[9];
  const float* Lm = (const float*)d_in[10];
  float* out = (float*)d_out;

  // Workspace layout (byte offsets, 256B aligned)
  char* w = (char*)d_ws;
  const size_t SZ_XB   = (size_t)BATCH * KX * 2;
  const size_t SZ_LB1  = (size_t)NL * KX * 2;
  const size_t SZ_WP   = (size_t)3 * 27 * NH * NH * 2;
  const size_t SZ_HP   = (size_t)VOX * BATCH * NH * 2;
  const size_t SZ_LIN1 = (size_t)BATCH * NL * 4;
  const size_t SZ_CONV = (size_t)VOX * BATCH * NH * 4;

  size_t off = 0;
  __bf16* xb   = (__bf16*)(w + off); off += SZ_XB;
  __bf16* Lb   = (__bf16*)(w + off); off += 3 * SZ_LB1;
  __bf16* Wp   = (__bf16*)(w + off); off += SZ_WP;
  __bf16* hp   = (__bf16*)(w + off); off += SZ_HP;
  float*  lin  = (float*)(w + off);  off += 3 * SZ_LIN1;
  float*  convZ= (float*)(w + off);  off += SZ_CONV;
  float*  convR= (float*)(w + off);  off += SZ_CONV;
  float*  zbuf = (float*)(w + off);  off += SZ_CONV;
  __bf16* rhp  = (__bf16*)(w + off); off += SZ_HP;
  float*  convM = convZ;  // convZ consumed by gate_zr before conv M runs

  const long NELT = (long)VOX * BATCH * NH;  // 33.5M

  // 1) pack to bf16
  pack_bf16<<<(BATCH * KX) / 256, 256, 0, stream>>>(x, xb, (long)BATCH * KX);
  pack_bf16<<<(NL * KX) / 256, 256, 0, stream>>>(Lz, Lb + 0 * (size_t)NL * KX, (long)NL * KX);
  pack_bf16<<<(NL * KX) / 256, 256, 0, stream>>>(Lr, Lb + 1 * (size_t)NL * KX, (long)NL * KX);
  pack_bf16<<<(NL * KX) / 256, 256, 0, stream>>>(Lm, Lb + 2 * (size_t)NL * KX, (long)NL * KX);
  pack_w<<<(27 * NH * NH + 255) / 256, 256, 0, stream>>>(Wz, Wp + 0 * (size_t)27 * NH * NH);
  pack_w<<<(27 * NH * NH + 255) / 256, 256, 0, stream>>>(Wr, Wp + 1 * (size_t)27 * NH * NH);
  pack_w<<<(27 * NH * NH + 255) / 256, 256, 0, stream>>>(Wm, Wp + 2 * (size_t)27 * NH * NH);
  pack_h<<<(unsigned)(NELT / 256), 256, 0, stream>>>(hidden, hp);

  // 2) linear projections: lin[g] = x @ L[g]^T   (M=4096, N=8192, K=1024)
  dim3 ggrid(NL / 128, BATCH / 64);   // (64, 64)
  gemm_bf16_lds<<<ggrid, 256, 0, stream>>>(xb, Lb + 0 * (size_t)NL * KX,
                                           lin + 0 * (size_t)BATCH * NL, BATCH, NL, KX);
  gemm_bf16_lds<<<ggrid, 256, 0, stream>>>(xb, Lb + 1 * (size_t)NL * KX,
                                           lin + 1 * (size_t)BATCH * NL, BATCH, NL, KX);
  gemm_bf16_lds<<<ggrid, 256, 0, stream>>>(xb, Lb + 2 * (size_t)NL * KX,
                                           lin + 2 * (size_t)BATCH * NL, BATCH, NL, KX);

  // 3) z / r convolutions
  dim3 cgrid(BATCH / 64, VOX);        // (64, 64)
  conv_gemm_lds<<<cgrid, 256, 0, stream>>>(hp, Wp + 0 * (size_t)27 * NH * NH, convZ);
  conv_gemm_lds<<<cgrid, 256, 0, stream>>>(hp, Wp + 1 * (size_t)27 * NH * NH, convR);

  // 4) gates: z, r; emit r*h bf16-packed for the m conv
  gate_zr<<<(unsigned)(NELT / 256), 256, 0, stream>>>(
      convZ, convR,
      lin + 0 * (size_t)BATCH * NL, lin + 1 * (size_t)BATCH * NL,
      bz, br, hidden, zbuf, rhp);

  // 5) m convolution on r*h
  conv_gemm_lds<<<cgrid, 256, 0, stream>>>(rhp, Wp + 2 * (size_t)27 * NH * NH, convM);

  // 6) output
  final_gate<<<(unsigned)(NELT / 256), 256, 0, stream>>>(
      convM, lin + 2 * (size_t)BATCH * NL, bm, hidden, zbuf, out);
}
